// MLAAttention_1846835938081
// MI455X (gfx1250) — matrix operand order repaired and checked by reading, not compiled, not run on hardware
//
#include <hip/hip_runtime.h>
#include <hip/hip_bf16.h>
#include <cstdint>

// ---------------------------------------------------------------------------
// MLA attention for MI455X (gfx1250): bf16 WMMA everywhere, f32 accumulate.
// Round 3: fix async-to-LDS builtin signature (takes generic int4* pointers).
// TDM (tensor_load_to_lds) staging + double-buffered GEMM k-loop retained.
// ---------------------------------------------------------------------------

typedef unsigned short u16;
typedef __attribute__((ext_vector_type(16))) __bf16        v16bf;
typedef __attribute__((ext_vector_type(8)))  float         v8f;
typedef __attribute__((ext_vector_type(4)))  unsigned int  u32x4;
typedef __attribute__((ext_vector_type(4)))  int           i32x4g;  // async-LDS op operand

union FragAB { v16bf v; u32x4 q[2]; };
union U4U16  { u32x4 q; u16 s[8]; };

#if __has_builtin(__builtin_amdgcn_tensor_load_to_lds)
#define HAS_TDM 1
#else
#define HAS_TDM 0
#endif
#if __has_builtin(__builtin_amdgcn_global_load_async_to_lds_b128)
#define HAS_ASYNC 1
#else
#define HAS_ASYNC 0
#endif
#if __has_include(<hip/amd_detail/amd_gfx1250_TDM.h>)
#define TDM_ARGS6 1
#else
#define TDM_ARGS6 0
#endif

__device__ __forceinline__ u16 f2bf(float f) {
  union { float f; uint32_t u; } c; c.f = f;
  uint32_t u = c.u;
  uint32_t r = (u + 0x7FFFu + ((u >> 16) & 1u)) >> 16;  // RNE
  return (u16)r;
}
__device__ __forceinline__ float bf2f(u16 h) {
  union { uint32_t u; float f; } c; c.u = ((uint32_t)h) << 16;
  return c.f;
}
__device__ __forceinline__ v8f wmma_bf16(v16bf a, v16bf b, v8f c) {
  return __builtin_amdgcn_wmma_f32_16x16x32_bf16(false, a, false, b,
                                                 (short)0, c, false, false);
}

// LDS byte offset of a generic pointer into shared memory
__device__ __forceinline__ uint32_t lds_off(const u16* p) {
  return (uint32_t)(unsigned long long)(__attribute__((address_space(3))) u16*)(u16*)p;
}

// Per-lane 16-byte global -> LDS copy (async if available)
__device__ __forceinline__ void cp16_async(u16* lds, const u16* g) {
#if HAS_ASYNC
  __builtin_amdgcn_global_load_async_to_lds_b128((i32x4g*)g, (i32x4g*)lds, 0, 0);
#else
  *(u32x4*)lds = *(const u32x4*)g;
#endif
}

// Wait for all outstanding stage traffic issued by this wave
__device__ __forceinline__ void wait_stage() {
#if HAS_TDM
#if __has_builtin(__builtin_amdgcn_s_wait_tensorcnt)
  __builtin_amdgcn_s_wait_tensorcnt(0);
#else
  asm volatile("s_wait_tensorcnt 0x0" ::: "memory");
#endif
#endif
#if HAS_ASYNC
#if __has_builtin(__builtin_amdgcn_s_wait_asynccnt)
  __builtin_amdgcn_s_wait_asynccnt(0);
#else
  asm volatile("s_wait_asynccnt 0x0" ::: "memory");
#endif
#endif
}

#if HAS_TDM
typedef __attribute__((ext_vector_type(4))) unsigned int tdm_u4;
typedef __attribute__((ext_vector_type(8))) int          tdm_i8;
typedef __attribute__((ext_vector_type(4))) int          tdm_i4;

// 2D tile DMA: tile_w elements (2B each) x tile_h rows, row stride in elems.
// D# layout per CDNA5 ISA ch.8: group0 {count, lds_addr, global_addr, type=2},
// group1 {data_size=2B, tensor dims, tile dims, dim0 stride}.
__device__ __forceinline__ void tdm_load_2d(const u16* gsrc, const u16* ldst,
                                            uint32_t tile_w, uint32_t tile_h,
                                            uint32_t tensor_w, uint32_t tensor_h,
                                            uint64_t stride_elems) {
  uint64_t ga = (uint64_t)(const void*)gsrc;
  tdm_u4 g0;
  g0[0] = 1u;                                           // count=1, user D#
  g0[1] = lds_off(ldst);                                // LDS byte address
  g0[2] = (uint32_t)ga;                                 // global_addr[31:0]
  g0[3] = (uint32_t)((ga >> 32) & 0x1FFFFFFu) | (2u << 30);  // [56:32] | type=2
  tdm_i8 g1;
  g1[0] = (int)(1u << 16);                              // data_size=1 -> 2 bytes
  g1[1] = (int)((tensor_w & 0xFFFFu) << 16);            // tensor_dim0[15:0]
  g1[2] = (int)((tensor_w >> 16) | ((tensor_h & 0xFFFFu) << 16));
  g1[3] = (int)((tensor_h >> 16) | (tile_w << 16));     // tile_dim0
  g1[4] = (int)tile_h;                                  // tile_dim1
  g1[5] = (int)(uint32_t)stride_elems;                  // dim0 stride lo
  g1[6] = (int)(uint32_t)(stride_elems >> 32);          // dim0 stride hi
  g1[7] = 0;
  tdm_i4 z4 = {0, 0, 0, 0};
#if TDM_ARGS6
  tdm_i8 z8 = {0, 0, 0, 0, 0, 0, 0, 0};
  __builtin_amdgcn_tensor_load_to_lds(g0, g1, z4, z4, z8, 0);
#else
  __builtin_amdgcn_tensor_load_to_lds(g0, g1, z4, z4, 0);
#endif
}
#endif  // HAS_TDM

// ---------------------------------------------------------------------------
// f32 -> bf16 conversion
// ---------------------------------------------------------------------------
__global__ __launch_bounds__(256)
void k_cvt(const float* __restrict__ in, u16* __restrict__ out, int n) {
  int i = blockIdx.x * 256 + threadIdx.x;
  if (i < n) out[i] = f2bf(in[i]);
}

// ---------------------------------------------------------------------------
// bf16 GEMM: C[M,N] = A[M,K] @ B[K,N], row-major, dims multiples of 128/128/32.
// 8 wave32 per block; wave = 32x64 via 2x4 WMMA tiles. Double-buffered k-loop:
// A tile staged by TDM (wave 0) or per-lane async-to-LDS; B tile transposed
// in-thread to Bs[n][k]; tile t+1 staged while tile t feeds the WMMAs.
// ---------------------------------------------------------------------------
__global__ __launch_bounds__(256)
void k_gemm_bf16(const u16* __restrict__ A, const u16* __restrict__ B,
                 float* __restrict__ Cf, u16* __restrict__ Cb,
                 int M, int N, int K) {
  __shared__ u16 As[2][128 * 32];   // [m][k]
  __shared__ u16 Bs[2][128 * 32];   // [n][k]

  const int tid   = threadIdx.x;
  const int lane  = tid & 31, wid = tid >> 5;
  const int wm    = wid & 3, wn = wid >> 2;
  const int mlane = lane & 15, kh = lane >> 4;
  const int bm    = blockIdx.y, bn = blockIdx.x;

  const int arow  = tid >> 1, acol = (tid & 1) << 4;
  const int bkrow = tid >> 3, bcol = (tid & 7) << 4;

  auto stageA = [&](int k0, int buf) {
#if HAS_TDM
    if (wid == 0)
      tdm_load_2d(A + (size_t)(bm * 128) * K + k0, &As[buf][0],
                  /*tile*/ 32, 128, /*tensor*/ (uint32_t)K, (uint32_t)M,
                  (uint64_t)K);
#else
    const u16* src = A + (size_t)(bm * 128 + arow) * K + k0 + acol;
    cp16_async(&As[buf][arow * 32 + acol], src);
    cp16_async(&As[buf][arow * 32 + acol + 8], src + 8);
#endif
  };
  auto stageB = [&](int k0, int buf) {
    const u16* src = B + (size_t)(k0 + bkrow) * N + bn * 128 + bcol;
    U4U16 t0, t1; t0.q = *(const u32x4*)src; t1.q = *(const u32x4*)(src + 8);
#pragma unroll
    for (int e = 0; e < 8; ++e) Bs[buf][(bcol + e) * 32 + bkrow]     = t0.s[e];
#pragma unroll
    for (int e = 0; e < 8; ++e) Bs[buf][(bcol + 8 + e) * 32 + bkrow] = t1.s[e];
  };

  v8f acc[2][4] = {};
  const int ktiles = K >> 5;

  stageA(0, 0);
  stageB(0, 0);

  for (int t = 0; t < ktiles; ++t) {
    const int cur = t & 1;
    wait_stage();
    __syncthreads();
    if (t + 1 < ktiles) {           // overlap next-tile staging with WMMAs
      stageA((t + 1) << 5, cur ^ 1);
      stageB((t + 1) << 5, cur ^ 1);
    }
    FragAB a[2], b[4];
#pragma unroll
    for (int i = 0; i < 2; ++i) {
      // A frag 16x32: lane chunks at K = kh*8 and 16+kh*8
      const u16* p = &As[cur][(wm * 32 + i * 16 + mlane) * 32 + kh * 8];
      a[i].q[0] = *(const u32x4*)p;
      a[i].q[1] = *(const u32x4*)(p + 16);
    }
#pragma unroll
    for (int j = 0; j < 4; ++j) {
      // B frag 32x16: lane chunks at K = kh*8 and 16+kh*8
      const u16* p = &Bs[cur][(wn * 64 + j * 16 + mlane) * 32 + kh * 8];
      b[j].q[0] = *(const u32x4*)p;
      b[j].q[1] = *(const u32x4*)(p + 16);
    }
#pragma unroll
    for (int i = 0; i < 2; ++i)
#pragma unroll
      for (int j = 0; j < 4; ++j)
        acc[i][j] = wmma_bf16(a[i].v, b[j].v, acc[i][j]);
  }

  const int mhalf = kh * 8;
#pragma unroll
  for (int i = 0; i < 2; ++i)
#pragma unroll
    for (int j = 0; j < 4; ++j) {
      int col = bn * 128 + wn * 64 + j * 16 + mlane;
#pragma unroll
      for (int r = 0; r < 8; ++r) {
        int row = bm * 128 + wm * 32 + i * 16 + mhalf + r;
        float v = acc[i][j][r];
        if (Cf) Cf[(size_t)row * N + col] = v;
        if (Cb) Cb[(size_t)row * N + col] = f2bf(v);
      }
    }
}

// ---------------------------------------------------------------------------
// Repack + RoPE kernels.  Qcat[b][h][s][192], Kcat[b][hkv][s][192],
// Vc[b][hkv][s][128]  (all bf16)
// ---------------------------------------------------------------------------
__global__ __launch_bounds__(256)
void k_repack_q(const u16* __restrict__ Qtmp, const u16* __restrict__ Qr,
                const int* __restrict__ pos, const float* __restrict__ cosc,
                const float* __restrict__ sinc, u16* __restrict__ Qcat) {
  size_t i = (size_t)blockIdx.x * 256 + threadIdx.x;
  if (i >= (size_t)32 * 2048 * 192) return;
  int d = (int)(i % 192); size_t t = i / 192;
  int s = (int)(t % 2048); size_t bh = t / 2048;
  int h = (int)(bh % 16); int b = (int)(bh / 16);
  size_t bs = (size_t)b * 2048 + s;
  u16 o;
  if (d < 128) {
    o = Qtmp[bs * 2048 + (size_t)h * 128 + d];
  } else {
    int dr = d - 128;
    int p  = pos[s];
    float c  = cosc[(size_t)p * 64 + dr];
    float sn = sinc[(size_t)p * 64 + dr];
    float x  = bf2f(Qr[bs * 1024 + (size_t)h * 64 + dr]);
    float xr = (dr < 32) ? -bf2f(Qr[bs * 1024 + (size_t)h * 64 + dr + 32])
                         :  bf2f(Qr[bs * 1024 + (size_t)h * 64 + dr - 32]);
    o = f2bf(x * c + xr * sn);
  }
  Qcat[i] = o;
}

__global__ __launch_bounds__(256)
void k_repack_k(const u16* __restrict__ Ktmp, const u16* __restrict__ Kr,
                const int* __restrict__ pos, const float* __restrict__ cosc,
                const float* __restrict__ sinc, u16* __restrict__ Kcat) {
  size_t i = (size_t)blockIdx.x * 256 + threadIdx.x;
  if (i >= (size_t)8 * 2048 * 192) return;
  int d = (int)(i % 192); size_t t = i / 192;
  int s = (int)(t % 2048); size_t bh = t / 2048;
  int h = (int)(bh % 4); int b = (int)(bh / 4);
  size_t bs = (size_t)b * 2048 + s;
  u16 o;
  if (d < 128) {
    o = Ktmp[bs * 512 + (size_t)h * 128 + d];
  } else {
    int dr = d - 128;
    int p  = pos[s];
    float c  = cosc[(size_t)p * 64 + dr];
    float sn = sinc[(size_t)p * 64 + dr];
    float x  = bf2f(Kr[bs * 256 + (size_t)h * 64 + dr]);
    float xr = (dr < 32) ? -bf2f(Kr[bs * 256 + (size_t)h * 64 + dr + 32])
                         :  bf2f(Kr[bs * 256 + (size_t)h * 64 + dr - 32]);
    o = f2bf(x * c + xr * sn);
  }
  Kcat[i] = o;
}

__global__ __launch_bounds__(256)
void k_repack_v(const u16* __restrict__ Vtmp, u16* __restrict__ Vc) {
  size_t i = (size_t)blockIdx.x * 256 + threadIdx.x;
  if (i >= (size_t)8 * 2048 * 128) return;
  int d = (int)(i % 128); size_t t = i / 128;
  int s = (int)(t % 2048); size_t bh = t / 2048;
  int hk = (int)(bh % 4); int b = (int)(bh / 4);
  Vc[i] = Vtmp[((size_t)b * 2048 + s) * 512 + (size_t)hk * 128 + d];
}

// ---------------------------------------------------------------------------
// Flash attention: block = (q-tile of 128) x (head) x (batch), 256 threads.
// Each wave owns 16 query rows; K/V streamed in 64-key LDS tiles; causal.
// Q/K staged via async-to-LDS when available.
// Dynamic LDS: Qs 128x192, Ks 64x192, Vts 128x64 ([dv][key]), Ps 8x16x64.
// ---------------------------------------------------------------------------
__global__ __launch_bounds__(256)
void k_attention(const u16* __restrict__ Qcat, const u16* __restrict__ Kcat,
                 const u16* __restrict__ Vc, u16* __restrict__ attn) {
  extern __shared__ u16 smem[];
  u16* Qs  = smem;           // 24576 u16
  u16* Ks  = smem + 24576;   // 12288 u16
  u16* Vts = smem + 36864;   //  8192 u16
  u16* Ps  = smem + 45056;   //  8192 u16

  const int tid   = threadIdx.x;
  const int lane  = tid & 31, w = tid >> 5;
  const int mlane = lane & 15, kh = lane >> 4, mhalf = kh * 8;
  const int q0 = blockIdx.x * 128;
  const int h  = blockIdx.y;
  const int b  = blockIdx.z;
  const int hk = h >> 2;   // G = NH_Q/NH_KV = 4

  // Stage Q tile once (async)
  {
    const u16* src = Qcat + ((size_t)(b * 16 + h) * 2048 + q0) * 192;
    for (int idx = tid; idx < 128 * 24; idx += 256) {
      int row = idx / 24, c = idx % 24;
      cp16_async(&Qs[row * 192 + c * 8], src + (size_t)row * 192 + c * 8);
    }
  }

  float row_m[8], row_l[8];
#pragma unroll
  for (int r = 0; r < 8; ++r) { row_m[r] = -3.0e38f; row_l[r] = 0.f; }
  v8f o[8] = {};

  const int ntiles = q0 / 64 + 2;            // keys 0 .. q0+127
  const float scale = 0.07216878364870322f;  // 1/sqrt(192)

  for (int kt = 0; kt < ntiles; ++kt) {
    const int k0 = kt * 64;
    __syncthreads();   // previous tile's compute done before overwrite
    // Stage K tile [key][feat] (async)
    {
      const u16* src = Kcat + ((size_t)(b * 4 + hk) * 2048 + k0) * 192;
      for (int idx = tid; idx < 64 * 24; idx += 256) {
        int row = idx / 24, c = idx % 24;
        cp16_async(&Ks[row * 192 + c * 8], src + (size_t)row * 192 + c * 8);
      }
    }
    // Stage V tile transposed -> Vts[dv][key]
    {
      const u16* src = Vc + ((size_t)(b * 4 + hk) * 2048 + k0) * 128;
      for (int idx = tid; idx < 64 * 16; idx += 256) {
        int key = idx >> 4, c = (idx & 15) * 8;
        U4U16 t; t.q = *(const u32x4*)(src + (size_t)key * 128 + c);
#pragma unroll
        for (int e = 0; e < 8; ++e) Vts[(c + e) * 64 + key] = t.s[e];
      }
    }
    wait_stage();
    __syncthreads();

    if (k0 <= q0 + w * 16 + 15) {   // wave has at least one unmasked key
      // ---- scores S = Q @ K^T (16q x 64k), K-dim 192 in 6 steps ----
      v8f sc[4] = {};
#pragma unroll
      for (int kk = 0; kk < 6; ++kk) {
        FragAB aq;
        const u16* pa = &Qs[(w * 16 + mlane) * 192 + kk * 32 + kh * 8];
        aq.q[0] = *(const u32x4*)pa; aq.q[1] = *(const u32x4*)(pa + 16);
#pragma unroll
        for (int j = 0; j < 4; ++j) {
          FragAB bk;
          const u16* pb = &Ks[(j * 16 + mlane) * 192 + kk * 32 + kh * 8];
          bk.q[0] = *(const u32x4*)pb; bk.q[1] = *(const u32x4*)(pb + 16);
          sc[j] = wmma_bf16(aq.v, bk.v, sc[j]);
        }
      }
      // ---- scale + causal mask + tile row-max ----
      float tm[8];
#pragma unroll
      for (int r = 0; r < 8; ++r) tm[r] = -3.0e38f;
#pragma unroll
      for (int j = 0; j < 4; ++j) {
        int key = k0 + j * 16 + mlane;
#pragma unroll
        for (int r = 0; r < 8; ++r) {
          int qi = q0 + w * 16 + mhalf + r;
          float s = sc[j][r] * scale;
          s = (key <= qi) ? s : -3.0e38f;
          sc[j][r] = s;
          tm[r] = fmaxf(tm[r], s);
        }
      }
      // ---- online softmax update (16-lane row reductions) ----
#pragma unroll
      for (int r = 0; r < 8; ++r) {
#pragma unroll
        for (int off = 1; off < 16; off <<= 1)
          tm[r] = fmaxf(tm[r], __shfl_xor(tm[r], off, 32));
        float nm = fmaxf(row_m[r], tm[r]);
        float al = __expf(row_m[r] - nm);
        row_m[r] = nm;
        row_l[r] *= al;
#pragma unroll
        for (int jj = 0; jj < 8; ++jj) o[jj][r] *= al;
      }
      // ---- P = exp(S - m), write bf16 to per-wave LDS scratch ----
      u16* PsW = &Ps[w * 1024];
      float ts[8] = {0, 0, 0, 0, 0, 0, 0, 0};
#pragma unroll
      for (int j = 0; j < 4; ++j) {
#pragma unroll
        for (int r = 0; r < 8; ++r) {
          float s = sc[j][r];
          float p = (s > -1.0e37f) ? __expf(s - row_m[r]) : 0.f;
          ts[r] += p;
          PsW[(mhalf + r) * 64 + j * 16 + mlane] = f2bf(p);
        }
      }
#pragma unroll
      for (int r = 0; r < 8; ++r) {
#pragma unroll
        for (int off = 1; off < 16; off <<= 1)
          ts[r] += __shfl_xor(ts[r], off, 32);
        row_l[r] += ts[r];
      }
      // ---- O += P @ V  (16q x 128dv, key-dim 64 in 2 steps) ----
#pragma unroll
      for (int kc = 0; kc < 2; ++kc) {
        FragAB ap;
        const u16* pp = &PsW[mlane * 64 + kc * 32 + kh * 8];
        ap.q[0] = *(const u32x4*)pp; ap.q[1] = *(const u32x4*)(pp + 16);
#pragma unroll
        for (int jj = 0; jj < 8; ++jj) {
          FragAB bv;
          const u16* pv = &Vts[(jj * 16 + mlane) * 64 + kc * 32 + kh * 8];
          bv.q[0] = *(const u32x4*)pv; bv.q[1] = *(const u32x4*)(pv + 16);
          o[jj] = wmma_bf16(ap.v, bv.v, o[jj]);
        }
      }
    }
  }
  // ---- normalize + store bf16 into attn[b*S+s][h*128+dv] ----
#pragma unroll
  for (int r = 0; r < 8; ++r) {
    float inv = row_l[r] > 0.f ? 1.f / row_l[r] : 0.f;
    int srow = q0 + w * 16 + mhalf + r;
    size_t base = ((size_t)b * 2048 + srow) * 2048 + (size_t)h * 128;
#pragma unroll
    for (int jj = 0; jj < 8; ++jj)
      attn[base + jj * 16 + mlane] = f2bf(o[jj][r] * inv);
  }
}

// ---------------------------------------------------------------------------
// Host orchestration
// ---------------------------------------------------------------------------
extern "C" void kernel_launch(void* const* d_in, const int* in_sizes, int n_in,
                              void* d_out, int out_size, void* d_ws, size_t ws_size,
                              hipStream_t stream) {
  (void)in_sizes; (void)n_in; (void)out_size; (void)ws_size;

  const float* x    = (const float*)d_in[0];
  const float* cosc = (const float*)d_in[1];
  const float* sinc = (const float*)d_in[2];
  const int*   pos  = (const int*)d_in[3];
  /* d_in[4] attn_mask: causal mask implemented analytically */
  const float* wdkv = (const float*)d_in[5];
  const float* wuk  = (const float*)d_in[6];
  const float* wuv  = (const float*)d_in[7];
  const float* wdq  = (const float*)d_in[8];
  const float* wuq  = (const float*)d_in[9];
  const float* wkr  = (const float*)d_in[10];
  const float* wqr  = (const float*)d_in[11];
  const float* wo   = (const float*)d_in[12];

  float* out     = (float*)d_out;                    // [4096, 2048]
  float* ckv_out = out + (size_t)4096 * 2048;        // [4096, 512]

  u16* ws = (u16*)d_ws;
  size_t off = 0;
  auto alloc = [&](size_t n) { u16* p = ws + off; off += n; return p; };
  u16* x_bf   = alloc((size_t)4096 * 2048);
  u16* wdkv_b = alloc((size_t)2048 * 512);
  u16* wuk_b  = alloc((size_t)512 * 512);
  u16* wuv_b  = alloc((size_t)512 * 512);
  u16* wdq_b  = alloc((size_t)2048 * 1536);
  u16* wuq_b  = alloc((size_t)1536 * 2048);
  u16* wkr_b  = alloc((size_t)2048 * 256);
  u16* wqr_b  = alloc((size_t)2048 * 1024);
  u16* wo_b   = alloc((size_t)2048 * 2048);
  u16* ckv_b  = alloc((size_t)4096 * 512);
  u16* cq_b   = alloc((size_t)4096 * 1536);
  u16* ktmp   = alloc((size_t)4096 * 512);
  u16* vtmp   = alloc((size_t)4096 * 512);
  u16* qtmp   = alloc((size_t)4096 * 2048);   // later reused as attn bf16
  u16* qr_t   = alloc((size_t)4096 * 1024);
  u16* kr_t   = alloc((size_t)4096 * 256);
  u16* qcat   = alloc((size_t)32 * 2048 * 192);
  u16* kcat   = alloc((size_t)8 * 2048 * 192);
  u16* vc     = alloc((size_t)8 * 2048 * 128);
  u16* attn_b = qtmp;  // qtmp fully consumed by repack_q before attention

  auto cvt = [&](const float* in, u16* o2, size_t n) {
    k_cvt<<<dim3((unsigned)((n + 255) / 256)), dim3(256), 0, stream>>>(in, o2, (int)n);
  };
  cvt(x,    x_bf,   (size_t)4096 * 2048);
  cvt(wdkv, wdkv_b, (size_t)2048 * 512);
  cvt(wuk,  wuk_b,  (size_t)512 * 512);
  cvt(wuv,  wuv_b,  (size_t)512 * 512);
  cvt(wdq,  wdq_b,  (size_t)2048 * 1536);
  cvt(wuq,  wuq_b,  (size_t)1536 * 2048);
  cvt(wkr,  wkr_b,  (size_t)2048 * 256);
  cvt(wqr,  wqr_b,  (size_t)2048 * 1024);
  cvt(wo,   wo_b,   (size_t)2048 * 2048);

  auto gemm = [&](const u16* A, const u16* B, float* Cf, u16* Cb,
                  int M, int N, int K) {
    k_gemm_bf16<<<dim3(N / 128, M / 128), dim3(256), 0, stream>>>(A, B, Cf, Cb, M, N, K);
  };

  // projections
  gemm(x_bf,  wdkv_b, ckv_out, ckv_b, 4096,  512, 2048);  // c_kv (f32 out + bf16)
  gemm(ckv_b, wuk_b,  nullptr, ktmp,  4096,  512,  512);  // K_nope
  gemm(ckv_b, wuv_b,  nullptr, vtmp,  4096,  512,  512);  // V
  gemm(x_bf,  wdq_b,  nullptr, cq_b,  4096, 1536, 2048);  // c_q
  gemm(cq_b,  wuq_b,  nullptr, qtmp,  4096, 2048, 1536);  // Q_nope
  gemm(x_bf,  wqr_b,  nullptr, qr_t,  4096, 1024, 2048);  // Q_rope
  gemm(x_bf,  wkr_b,  nullptr, kr_t,  4096,  256, 2048);  // K_rope

  // RoPE + repack to attention layouts
  k_repack_q<<<dim3(49152), dim3(256), 0, stream>>>(qtmp, qr_t, pos, cosc, sinc, qcat);
  k_repack_k<<<dim3(12288), dim3(256), 0, stream>>>(ktmp, kr_t, pos, cosc, sinc, kcat);
  k_repack_v<<<dim3(8192),  dim3(256), 0, stream>>>(vtmp, vc);

  // flash attention: grid = (S/128, NH_Q, B), 104 KB dynamic LDS
  k_attention<<<dim3(16, 16, 2), dim3(256), 106496, stream>>>(qcat, kcat, vc, attn_b);

  // output projection
  gemm(attn_b, wo_b, out, nullptr, 4096, 2048, 2048);
}
